// Qwen3Attention_25855703122387
// MI455X (gfx1250) — matrix operand order repaired and checked by reading, not compiled, not run on hardware
//
#include <hip/hip_runtime.h>
#include <hip/hip_bf16.h>
#include <stdint.h>

typedef __attribute__((ext_vector_type(16))) _Float16 v16h;
typedef __attribute__((ext_vector_type(8)))  float    v8f;

#define HIDDEN   1024
#define Q_OUT    2048
#define KV_OUT   1024
#define HEAD_DIM 128
#define N_Q      16
#define N_KV     8
#define SEQ      2048
#define BATCH    2
#define ROWS     (BATCH*SEQ)            // 4096
#define QKV_N    (Q_OUT + 2*KV_OUT)     // 4096

// async copy of 16 bytes global -> LDS (gfx1250 ASYNCcnt-tracked path)
__device__ __forceinline__ void async_b128(unsigned lds_off, unsigned long long gaddr) {
    asm volatile("global_load_async_to_lds_b128 %0, %1, off"
                 :: "v"(lds_off), "v"(gaddr) : "memory");
}
__device__ __forceinline__ void wait_async0() {
    asm volatile("s_wait_asynccnt 0x0" ::: "memory");
}

// ---------------------------------------------------------------------------
// Stage 0: fp32 -> f16 convert (optionally packing columns into a wider dst)
// ---------------------------------------------------------------------------
__global__ void cvt_pack_f16(const float* __restrict__ src, _Float16* __restrict__ dst,
                             int rows, int cols, int dstStride, int colOff) {
    size_t i = (size_t)blockIdx.x * blockDim.x + threadIdx.x;
    if (i >= (size_t)rows * cols) return;
    int r = (int)(i / cols), c = (int)(i % cols);
    dst[(size_t)r * dstStride + colOff + c] = (_Float16)src[i];
}

// ---------------------------------------------------------------------------
// f16 GEMM: C[M,N](f32) = A[M,K] @ B[K,N].  4 waves, 64x64 block tile,
// double-buffered LDS: async-stage A(k+1) and global-load B(k+1) BEFORE
// computing tile k, so the copy latency hides under the 16 WMMAs.
// B stored transposed so operand K-pairs merge into ds_load_b128.
// ---------------------------------------------------------------------------
__global__ void gemm_f16_wmma(const _Float16* __restrict__ A,
                              const _Float16* __restrict__ B,
                              float* __restrict__ C, int M, int N, int K) {
    __shared__ _Float16 sA[2][64 * 32];     // [m][k]  2 x 4 KB
    __shared__ _Float16 sBt[2][64 * 32];    // [n][k]  2 x 4 KB (transposed)

    const int tid  = threadIdx.x;        // 0..127
    const int lane = tid & 31;
    const int wid  = tid >> 5;
    const int col  = lane & 15;
    const int g    = lane >> 4;
    const int m0   = blockIdx.y * 64;
    const int n0   = blockIdx.x * 64;

    v8f acc[4];
    #pragma unroll
    for (int ns = 0; ns < 4; ++ns) acc[ns] = (v8f){};

    const unsigned sAb0 = (unsigned)(uintptr_t)&sA[0][0];
    const unsigned sAb1 = (unsigned)(uintptr_t)&sA[1][0];
    const int arow = tid >> 1, aseg = (tid & 1) * 16;   // A: 64 rows x 32 halves
    const int brow = tid >> 2, bseg = (tid & 3) * 16;   // B: 32 rows x 64 halves
    const unsigned aloff = (unsigned)(arow * 32 + aseg) * 2u;

    // ---- prologue: stage tile k0 = 0 into buffer 0 ----
    {
        unsigned long long ga = (unsigned long long)(uintptr_t)
            (A + (size_t)(m0 + arow) * K + aseg);
        async_b128(sAb0 + aloff, ga);
        async_b128(sAb0 + aloff + 16u, ga + 16ull);

        const float4* bp = reinterpret_cast<const float4*>(
            B + (size_t)brow * N + n0 + bseg);
        _Float16 btmp[16];
        *reinterpret_cast<float4*>(btmp)     = bp[0];
        *reinterpret_cast<float4*>(btmp + 8) = bp[1];
        #pragma unroll
        for (int j = 0; j < 16; ++j)
            sBt[0][(bseg + j) * 32 + brow] = btmp[j];
    }
    wait_async0();
    __syncthreads();

    int cur = 0;
    for (int k0 = 0; k0 < K; k0 += 32) {
        const int  nxt  = cur ^ 1;
        const bool more = (k0 + 32) < K;

        // ---- issue next tile's copies before computing this one ----
        _Float16 btmp[16];
        if (more) {
            unsigned long long ga = (unsigned long long)(uintptr_t)
                (A + (size_t)(m0 + arow) * K + (k0 + 32) + aseg);
            unsigned base = nxt ? sAb1 : sAb0;
            async_b128(base + aloff, ga);
            async_b128(base + aloff + 16u, ga + 16ull);

            const float4* bp = reinterpret_cast<const float4*>(
                B + (size_t)(k0 + 32 + brow) * N + n0 + bseg);
            *reinterpret_cast<float4*>(btmp)     = bp[0];
            *reinterpret_cast<float4*>(btmp + 8) = bp[1];

            __builtin_prefetch(A + (size_t)(m0 + arow) * K + (k0 + 64) + aseg, 0, 1);
            __builtin_prefetch(B + (size_t)(k0 + 64 + brow) * N + n0 + bseg, 0, 1);
        }

        // ---- compute tile k0 from buffer `cur` ----
        v16h a;
        #pragma unroll
        for (int v = 0; v < 8; ++v) {
            int kb = ((v < 4) ? 0 : 16) + g * 8 + (v & 3) * 2;
            a[2 * v]     = sA[cur][(wid * 16 + col) * 32 + kb];
            a[2 * v + 1] = sA[cur][(wid * 16 + col) * 32 + kb + 1];
        }
        #pragma unroll
        for (int ns = 0; ns < 4; ++ns) {
            v16h b;
            #pragma unroll
            for (int v = 0; v < 8; ++v) {
                int kr = ((v < 4) ? 0 : 16) + g * 8 + (v & 3) * 2;
                b[2 * v]     = sBt[cur][(ns * 16 + col) * 32 + kr];
                b[2 * v + 1] = sBt[cur][(ns * 16 + col) * 32 + kr + 1];
            }
            acc[ns] = __builtin_amdgcn_wmma_f32_16x16x32_f16(false, a, false, b,
                                                             (short)0, acc[ns],
                                                             false, false);
        }

        // ---- finish staging next tile ----
        if (more) {
            #pragma unroll
            for (int j = 0; j < 16; ++j)
                sBt[nxt][(bseg + j) * 32 + brow] = btmp[j];
        }
        wait_async0();
        __syncthreads();
        cur = nxt;
    }

    #pragma unroll
    for (int ns = 0; ns < 4; ++ns)
        #pragma unroll
        for (int r = 0; r < 8; ++r)
            C[(size_t)(m0 + wid * 16 + r + 8 * g) * N + n0 + ns * 16 + col] = acc[ns][r];
}

// ---------------------------------------------------------------------------
// Stage 2 epilogue: per-head RMSNorm + RoPE on q/k; v converted to f16.
// ---------------------------------------------------------------------------
__global__ void qk_epilogue(const float* __restrict__ qkv,
                            const float* __restrict__ qw,
                            const float* __restrict__ kw,
                            _Float16* __restrict__ qh,
                            _Float16* __restrict__ kh,
                            _Float16* __restrict__ vh) {
    const int slot = blockIdx.x;        // 0..31 head slots
    const int rowi = blockIdx.y;        // 0..4095
    const int d    = threadIdx.x;       // 0..127
    const int b    = rowi >> 11;
    const int t    = rowi & (SEQ - 1);

    float x = qkv[(size_t)rowi * QKV_N + slot * HEAD_DIM + d];

    if (slot >= 24) {                   // v head: plain convert
        int h = slot - 24;
        vh[(((size_t)(b * N_KV + h) * SEQ) + t) * HEAD_DIM + d] = (_Float16)x;
        return;
    }

    __shared__ float red[HEAD_DIM];
    __shared__ float xn[HEAD_DIM];
    red[d] = x * x;
    __syncthreads();
    #pragma unroll
    for (int off = 64; off > 0; off >>= 1) {
        if (d < off) red[d] += red[d + off];
        __syncthreads();
    }
    float ms  = red[0] * (1.0f / HEAD_DIM);
    float inv = rsqrtf(ms + 1e-6f);
    const float* w = (slot < 16) ? qw : kw;
    float vn = x * inv * w[d];
    xn[d] = vn;
    __syncthreads();

    float partner = (d < 64) ? -xn[d + 64] : xn[d - 64];
    int   fi   = d & 63;
    float freq = __expf(-(float)fi * (9.210340371976184f / 64.0f)); // 10000^(-fi/64)
    float ang  = (float)t * freq;
    float c = cosf(ang), s = sinf(ang);
    float outv = vn * c + partner * s;

    if (slot < 16)
        qh[(((size_t)(b * N_Q + slot) * SEQ) + t) * HEAD_DIM + d] = (_Float16)outv;
    else
        kh[(((size_t)(b * N_KV + (slot - 16)) * SEQ) + t) * HEAD_DIM + d] = (_Float16)outv;
}

// ---------------------------------------------------------------------------
// Stage 3: flash attention.  4 waves / 64 queries per block.  K async-staged,
// V staged transposed ([d][key] -> B-operand pairs merge into ds_load_b128),
// both double-buffered so chunk ck+1 copies overlap chunk ck's 16 WMMAs per wave.
// ---------------------------------------------------------------------------
__global__ void flash_attn_wmma(const _Float16* __restrict__ qh,
                                const _Float16* __restrict__ kh,
                                const _Float16* __restrict__ vh,
                                _Float16* __restrict__ oh) {
    const int tid  = threadIdx.x;       // 0..127
    const int lane = tid & 31;
    const int wid  = tid >> 5;
    const int col  = lane & 15;
    const int g    = lane >> 4;
    const int h    = blockIdx.y;
    const int b    = blockIdx.z;
    const int kvh  = h >> 1;            // GQA rep = 2
    const int qt   = blockIdx.x * 4 + wid;   // this wave's 16-query tile
    const float SCALE = 0.08838834764831845f;   // 1/sqrt(128)

    __shared__ _Float16 sQ[64 * HEAD_DIM];          // 16 KB
    __shared__ _Float16 sK[2][32 * HEAD_DIM];       // 16 KB
    __shared__ _Float16 sVt[2][HEAD_DIM * 32];      // 16 KB  [d][key]
    __shared__ _Float16 sP[4][16 * 32];             //  4 KB (per-wave)

    const unsigned sQb  = (unsigned)(uintptr_t)sQ;
    const unsigned sKb0 = (unsigned)(uintptr_t)&sK[0][0];
    const unsigned sKb1 = (unsigned)(uintptr_t)&sK[1][0];

    const int nchunk = 2 * blockIdx.x + 2;   // covers keys 0 .. q0+63
    const _Float16* kbase = kh + ((size_t)(b * N_KV + kvh) * SEQ) * HEAD_DIM;
    const _Float16* vbase = vh + ((size_t)(b * N_KV + kvh) * SEQ) * HEAD_DIM;

    const int vrow = tid >> 2;              // key row 0..31
    const int vseg = (tid & 3) * 32;        // d offset 0,32,64,96

    // ---- prologue: async Q tile + async K chunk0; V chunk0 load+transpose ----
    {
        const _Float16* qbase = qh +
            (((size_t)(b * N_Q + h) * SEQ) + (size_t)blockIdx.x * 64) * HEAD_DIM;
        unsigned long long gq = (unsigned long long)(uintptr_t)qbase;
        #pragma unroll
        for (int i = 0; i < 8; ++i) {
            unsigned off = (unsigned)(tid + i * 128) * 16u;
            async_b128(sQb + off, gq + off);
        }
        unsigned long long gk = (unsigned long long)(uintptr_t)kbase;
        #pragma unroll
        for (int i = 0; i < 4; ++i) {
            unsigned off = (unsigned)(tid + i * 128) * 16u;
            async_b128(sKb0 + off, gk + off);
        }
        const float4* vp = reinterpret_cast<const float4*>(
            vbase + (size_t)vrow * HEAD_DIM + vseg);
        _Float16 vtmp[32];
        #pragma unroll
        for (int i = 0; i < 4; ++i)
            *reinterpret_cast<float4*>(vtmp + 8 * i) = vp[i];
        #pragma unroll
        for (int j = 0; j < 32; ++j)
            sVt[0][(vseg + j) * 32 + vrow] = vtmp[j];
    }
    wait_async0();
    __syncthreads();

    // this wave's Q A-operands for the 4 D-chunks of 32
    v16h qa[4];
    #pragma unroll
    for (int c = 0; c < 4; ++c)
        #pragma unroll
        for (int v = 0; v < 8; ++v) {
            int kb = c * 32 + ((v < 4) ? 0 : 16) + g * 8 + (v & 3) * 2;
            qa[c][2 * v]     = sQ[(wid * 16 + col) * HEAD_DIM + kb];
            qa[c][2 * v + 1] = sQ[(wid * 16 + col) * HEAD_DIM + kb + 1];
        }

    float m_run[8], l_run[8];
    v8f   o[8];
    #pragma unroll
    for (int r = 0; r < 8; ++r) { m_run[r] = -1e30f; l_run[r] = 0.0f; }
    #pragma unroll
    for (int ds = 0; ds < 8; ++ds) o[ds] = (v8f){};

    int cur = 0;
    for (int ck = 0; ck < nchunk; ++ck) {
        const int  j0   = ck * 32;
        const int  nxt  = cur ^ 1;
        const bool more = (ck + 1) < nchunk;

        // ---- issue chunk ck+1 copies before computing chunk ck ----
        _Float16 vtmp[32];
        if (more) {
            unsigned long long gk = (unsigned long long)(uintptr_t)
                (kbase + (size_t)(j0 + 32) * HEAD_DIM);
            unsigned kb_ = nxt ? sKb1 : sKb0;
            #pragma unroll
            for (int i = 0; i < 4; ++i) {
                unsigned off = (unsigned)(tid + i * 128) * 16u;
                async_b128(kb_ + off, gk + off);
            }
            const float4* vp = reinterpret_cast<const float4*>(
                vbase + (size_t)(j0 + 32 + vrow) * HEAD_DIM + vseg);
            #pragma unroll
            for (int i = 0; i < 4; ++i)
                *reinterpret_cast<float4*>(vtmp + 8 * i) = vp[i];
        }

        // ---- S = Q K^T on buffer `cur` ----
        float sv[2][8];
        #pragma unroll
        for (int sub = 0; sub < 2; ++sub) {
            v8f s = {};
            #pragma unroll
            for (int c = 0; c < 4; ++c) {
                v16h bb;
                #pragma unroll
                for (int v = 0; v < 8; ++v) {
                    int d = c * 32 + ((v < 4) ? 0 : 16) + g * 8 + (v & 3) * 2;
                    bb[2 * v]     = sK[cur][(sub * 16 + col) * HEAD_DIM + d];
                    bb[2 * v + 1] = sK[cur][(sub * 16 + col) * HEAD_DIM + d + 1];
                }
                s = __builtin_amdgcn_wmma_f32_16x16x32_f16(false, qa[c], false, bb,
                                                           (short)0, s, false, false);
            }
            int kidx = j0 + sub * 16 + col;
            #pragma unroll
            for (int r = 0; r < 8; ++r) {
                int qidx = qt * 16 + r + 8 * g;
                sv[sub][r] = (kidx > qidx) ? -1e30f : s[r] * SCALE;
            }
        }

        // ---- online softmax (row lives in 16 lanes of this half) ----
        float rmax[8];
        #pragma unroll
        for (int r = 0; r < 8; ++r) rmax[r] = fmaxf(sv[0][r], sv[1][r]);
        #pragma unroll
        for (int off = 1; off < 16; off <<= 1)
            #pragma unroll
            for (int r = 0; r < 8; ++r)
                rmax[r] = fmaxf(rmax[r], __shfl_xor(rmax[r], off, 32));

        float alpha[8], rsum[8];
        #pragma unroll
        for (int r = 0; r < 8; ++r) {
            float mnew = fmaxf(m_run[r], rmax[r]);
            alpha[r] = __expf(m_run[r] - mnew);
            m_run[r] = mnew;
            rsum[r]  = 0.0f;
        }
        #pragma unroll
        for (int sub = 0; sub < 2; ++sub)
            #pragma unroll
            for (int r = 0; r < 8; ++r) {
                float p = __expf(sv[sub][r] - m_run[r]);
                sP[wid][(r + 8 * g) * 32 + sub * 16 + col] = (_Float16)p;
                rsum[r] += p;
            }
        #pragma unroll
        for (int off = 1; off < 16; off <<= 1)
            #pragma unroll
            for (int r = 0; r < 8; ++r)
                rsum[r] += __shfl_xor(rsum[r], off, 32);
        #pragma unroll
        for (int r = 0; r < 8; ++r) l_run[r] = l_run[r] * alpha[r] + rsum[r];
        #pragma unroll
        for (int ds = 0; ds < 8; ++ds)
            #pragma unroll
            for (int r = 0; r < 8; ++r) o[ds][r] *= alpha[r];

        // ---- P as A-operand (per-wave LDS region; same-wave DS ordering) ----
        v16h pa;
        #pragma unroll
        for (int v = 0; v < 8; ++v) {
            int kb = ((v < 4) ? 0 : 16) + g * 8 + (v & 3) * 2;
            pa[2 * v]     = sP[wid][col * 32 + kb];
            pa[2 * v + 1] = sP[wid][col * 32 + kb + 1];
        }
        // ---- O += P V : V-operand pairs contiguous in sVt -> ds_load_b128 ----
        #pragma unroll
        for (int ds = 0; ds < 8; ++ds) {
            v16h bb;
            #pragma unroll
            for (int v = 0; v < 8; ++v) {
                int kk = ((v < 4) ? 0 : 16) + g * 8 + (v & 3) * 2;
                bb[2 * v]     = sVt[cur][(ds * 16 + col) * 32 + kk];
                bb[2 * v + 1] = sVt[cur][(ds * 16 + col) * 32 + kk + 1];
            }
            o[ds] = __builtin_amdgcn_wmma_f32_16x16x32_f16(false, pa, false, bb,
                                                           (short)0, o[ds],
                                                           false, false);
        }

        // ---- finish staging chunk ck+1 ----
        if (more) {
            #pragma unroll
            for (int j = 0; j < 32; ++j)
                sVt[nxt][(vseg + j) * 32 + vrow] = vtmp[j];
        }
        wait_async0();
        __syncthreads();
        cur = nxt;
    }

    // normalize and write out: (B, T, Hq*D) f16
    #pragma unroll
    for (int ds = 0; ds < 8; ++ds)
        #pragma unroll
        for (int r = 0; r < 8; ++r) {
            size_t orow = (size_t)(b * SEQ + qt * 16 + r + 8 * g);
            oh[orow * Q_OUT + h * HEAD_DIM + ds * 16 + col] =
                (_Float16)(o[ds][r] / l_run[r]);
        }
}

// ---------------------------------------------------------------------------
// Host-side launcher
// ---------------------------------------------------------------------------
extern "C" void kernel_launch(void* const* d_in, const int* in_sizes, int n_in,
                              void* d_out, int out_size, void* d_ws, size_t ws_size,
                              hipStream_t stream) {
    const float* x  = (const float*)d_in[0];
    const float* Wq = (const float*)d_in[1];
    const float* Wk = (const float*)d_in[2];
    const float* Wv = (const float*)d_in[3];
    const float* Wo = (const float*)d_in[4];
    const float* qw = (const float*)d_in[5];
    const float* kw = (const float*)d_in[6];
    float* out = (float*)d_out;

    char* ws = (char*)d_ws;
    _Float16* xh   = (_Float16*)(ws + 0);                    //  8 MB  4096x1024
    _Float16* wqkv = (_Float16*)(ws + 8388608);              //  8 MB  1024x4096
    _Float16* woH  = (_Float16*)(ws + 16777216);             //  4 MB  2048x1024
    float*    qkv  = (float*)   (ws + 20971520);             // 64 MB  4096x4096
    _Float16* qh   = (_Float16*)(ws + 88080384);             // 16 MB  (B,Hq,T,D)
    _Float16* khp  = (_Float16*)(ws + 104857600);            //  8 MB  (B,Hkv,T,D)
    _Float16* vhp  = (_Float16*)(ws + 113246208);            //  8 MB  (B,Hkv,T,D)
    _Float16* ah   = (_Float16*)(ws + 121634816);            // 16 MB  4096x2048

    auto blocks = [](size_t n) { return (unsigned)((n + 255) / 256); };

    // Stage 0: converts / packing
    cvt_pack_f16<<<blocks((size_t)ROWS * HIDDEN), 256, 0, stream>>>(x, xh, ROWS, HIDDEN, HIDDEN, 0);
    cvt_pack_f16<<<blocks((size_t)HIDDEN * Q_OUT), 256, 0, stream>>>(Wq, wqkv, HIDDEN, Q_OUT, QKV_N, 0);
    cvt_pack_f16<<<blocks((size_t)HIDDEN * KV_OUT), 256, 0, stream>>>(Wk, wqkv, HIDDEN, KV_OUT, QKV_N, Q_OUT);
    cvt_pack_f16<<<blocks((size_t)HIDDEN * KV_OUT), 256, 0, stream>>>(Wv, wqkv, HIDDEN, KV_OUT, QKV_N, Q_OUT + KV_OUT);
    cvt_pack_f16<<<blocks((size_t)Q_OUT * HIDDEN), 256, 0, stream>>>(Wo, woH, Q_OUT, HIDDEN, HIDDEN, 0);

    // Stage 1: fused QKV projection  qkv = xh @ wqkv   (4096x4096, K=1024)
    gemm_f16_wmma<<<dim3(QKV_N / 64, ROWS / 64), 128, 0, stream>>>(xh, wqkv, qkv, ROWS, QKV_N, HIDDEN);

    // Stage 2: RMSNorm + RoPE epilogue -> qh/kh/vh
    qk_epilogue<<<dim3(32, ROWS), HEAD_DIM, 0, stream>>>(qkv, qw, kw, qh, khp, vhp);

    // Stage 3: flash attention -> ah (B,T,Hq*D)
    flash_attn_wmma<<<dim3(SEQ / 64, N_Q, BATCH), 128, 0, stream>>>(qh, khp, vhp, ah);

    // Stage 4: output projection  out = ah @ woH  (4096x1024, K=2048)
    gemm_f16_wmma<<<dim3(HIDDEN / 64, ROWS / 64), 128, 0, stream>>>(ah, woH, out, ROWS, HIDDEN, Q_OUT);
}
